// Autoformer_EncoderOnly_86861418594331
// MI455X (gfx1250) — compile-verified, matrix operand order checked by reading, not run on hardware
//
#include <hip/hip_runtime.h>
#include <math.h>

// ---- problem constants (match reference) ----
#define B_    64
#define L_    512
#define C_    32
#define D_    512
#define DFF_  2048
#define HS_   512
#define EL_   2
#define MA_   25
#define PAD_  ((MA_ - 1) / 2)
#define TOPK_ 6          // int(log(512)) = 6
#define EPS_  1e-5f

// ---- WMMA types (gfx1250, wave32) ----
typedef __attribute__((ext_vector_type(16))) __bf16 v16bf;
typedef __attribute__((ext_vector_type(8)))  float  v8f;

union Frag16 { v16bf v; unsigned int u[8]; };

__device__ __forceinline__ unsigned short f32_to_bf16(float f) {
  unsigned int u = __float_as_uint(f);
  unsigned int r = (u + 0x7FFFu + ((u >> 16) & 1u)) >> 16;  // RNE
  return (unsigned short)r;
}

__device__ __forceinline__ float gelu_exact(float x) {
  return 0.5f * x * (1.0f + erff(x * 0.70710678118654752f));
}

// =====================================================================
// Batched GEMM: C[z][M,N] = act(A[z][M,K] * op(W[z]) + bias[N])
//   bTrans=0: W is [K,N] row-major;  bTrans=1: W is [N,K] row-major (B^T path,
//   used for Q*K^T — staging is then K-contiguous and fully coalesced).
// block = 256 threads (8 wave32), tile 128x128, K step 32.
// Wave tiling: 4x2 grid of waves, each wave owns a 32x64 C tile
//   -> 2 A-frags x 4 B-frags, 8 WMMAs per K-chunk, 48 uint LDS loads
//      (vs 72 for a 16x128 strip): 1.5x less LDS traffic per WMMA.
// LDS tiles stored M-/N-major with K contiguous, +2 pad (stride 68B, odd bank
// count) so every fragment VGPR pair is one aligned 32-bit ds load.
// =====================================================================
#define TM 128
#define TN 128
#define TK 32
#define LDT 34   // padded LDS row stride in bf16 elements (even -> 4B aligned pairs)

__global__ __launch_bounds__(256)
void gemm_bf16_wmma(const float* __restrict__ A, const float* __restrict__ W,
                    const float* __restrict__ bias, float* __restrict__ Cout,
                    int M, int N, int K, int act /*0=none,1=gelu*/, int bTrans,
                    size_t strideA, size_t strideW, size_t strideC) {
  __shared__ __align__(16) unsigned short As[TM * LDT];  // [m][k]
  __shared__ __align__(16) unsigned short Bs[TN * LDT];  // [n][k]

  A    += (size_t)blockIdx.z * strideA;
  W    += (size_t)blockIdx.z * strideW;
  Cout += (size_t)blockIdx.z * strideC;

  const int tid    = threadIdx.x;
  const int lane   = tid & 31;
  const int wave   = tid >> 5;
  const int wm     = wave & 3;        // 0..3 -> 32-row group
  const int wn     = wave >> 2;       // 0..1 -> 64-col group
  const int laneLo = lane & 15;
  const int laneHi = lane >> 4;

  const int mBlock = blockIdx.y * TM;
  const int nBlock = blockIdx.x * TN;

  v8f acc[2][4];
#pragma unroll
  for (int am = 0; am < 2; ++am)
#pragma unroll
    for (int bn = 0; bn < 4; ++bn)
      acc[am][bn] = (v8f){0.f,0.f,0.f,0.f,0.f,0.f,0.f,0.f};

  for (int k0 = 0; k0 < K; k0 += TK) {
    // stage A tile 128x32 (f32 -> bf16), coalesced along K
#pragma unroll
    for (int i = 0; i < 16; ++i) {
      int idx = tid + i * 256;           // 0..4095
      int r = idx >> 5, c = idx & 31;
      int gr = mBlock + r;
      float v = (gr < M) ? A[(size_t)gr * K + (k0 + c)] : 0.f;
      As[r * LDT + c] = f32_to_bf16(v);
    }
    // stage W tile into Bs[n][k]
    if (bTrans == 0) {
#pragma unroll
      for (int i = 0; i < 16; ++i) {
        int idx = tid + i * 256;
        int r = idx >> 7, c = idx & 127;   // r=k, c=n (coalesced along N)
        int gn = nBlock + c;
        float v = (gn < N) ? W[(size_t)(k0 + r) * N + gn] : 0.f;
        Bs[c * LDT + r] = f32_to_bf16(v);
      }
    } else {
#pragma unroll
      for (int i = 0; i < 16; ++i) {
        int idx = tid + i * 256;
        int n = idx >> 5, c = idx & 31;    // coalesced along K
        int gn = nBlock + n;
        float v = (gn < N) ? W[(size_t)gn * K + (k0 + c)] : 0.f;
        Bs[n * LDT + c] = f32_to_bf16(v);
      }
    }
    __syncthreads();

    // two A fragments for this wave's 32 rows (ISA 16-bit 16x32 A layout)
    Frag16 af[2];
#pragma unroll
    for (int am = 0; am < 2; ++am) {
      const unsigned short* arow = &As[(wm * 32 + am * 16 + laneLo) * LDT];
#pragma unroll
      for (int j = 0; j < 8; ++j) {
        int kb = ((j < 4) ? (2 * j) : (16 + 2 * (j - 4))) + (laneHi ? 8 : 0);
        af[am].u[j] = *(const unsigned int*)&arow[kb];
      }
    }
    // 4 B fragments across the wave's 64 columns; 2 WMMAs each (A reuse)
#pragma unroll
    for (int bn = 0; bn < 4; ++bn) {
      Frag16 bfr;
      const unsigned short* brow = &Bs[(wn * 64 + bn * 16 + laneLo) * LDT];
#pragma unroll
      for (int j = 0; j < 8; ++j) {
        int kb = ((j < 4) ? (2 * j) : (16 + 2 * (j - 4))) + (laneHi ? 8 : 0);
        bfr.u[j] = *(const unsigned int*)&brow[kb];
      }
      acc[0][bn] = __builtin_amdgcn_wmma_f32_16x16x32_bf16(
          false, af[0].v, false, bfr.v, (short)0, acc[0][bn], false, false);
      acc[1][bn] = __builtin_amdgcn_wmma_f32_16x16x32_bf16(
          false, af[1].v, false, bfr.v, (short)0, acc[1][bn], false, false);
    }
    __syncthreads();
  }

  // epilogue: C/D layout — VGPR r -> row r (+8 for lanes 16..31), lane&15 -> col
#pragma unroll
  for (int bn = 0; bn < 4; ++bn) {
    int col = nBlock + wn * 64 + bn * 16 + laneLo;
    if (col >= N) continue;
    float bv = bias ? bias[col] : 0.f;
#pragma unroll
    for (int am = 0; am < 2; ++am) {
      int rowBase = mBlock + wm * 32 + am * 16 + (laneHi ? 8 : 0);
#pragma unroll
      for (int r = 0; r < 8; ++r) {
        int row = rowBase + r;
        if (row < M) {
          float v = acc[am][bn][r] + bv;
          if (act == 1) v = gelu_exact(v);
          Cout[(size_t)row * N + col] = v;
        }
      }
    }
  }
}

// =====================================================================
// LayerNorm over last dim n (one block / row)
// =====================================================================
__global__ void ln_kernel(const float* __restrict__ in, float* __restrict__ out,
                          const float* __restrict__ g, const float* __restrict__ bb,
                          int n) {
  const int row = blockIdx.x, tid = threadIdx.x;
  const float* x = in + (size_t)row * n;
  __shared__ float red[256];
  __shared__ float mv[2];
  float s = 0.f;
  for (int i = tid; i < n; i += 256) s += x[i];
  red[tid] = s; __syncthreads();
  for (int st = 128; st > 0; st >>= 1) { if (tid < st) red[tid] += red[tid + st]; __syncthreads(); }
  if (tid == 0) mv[0] = red[0] / (float)n;
  __syncthreads();
  const float m = mv[0];
  float v = 0.f;
  for (int i = tid; i < n; i += 256) { float d = x[i] - m; v += d * d; }
  red[tid] = v; __syncthreads();
  for (int st = 128; st > 0; st >>= 1) { if (tid < st) red[tid] += red[tid + st]; __syncthreads(); }
  if (tid == 0) mv[1] = rsqrtf(red[0] / (float)n + EPS_);
  __syncthreads();
  const float rs = mv[1];
  float* o = out + (size_t)row * n;
  for (int i = tid; i < n; i += 256) o[i] = (x[i] - m) * rs * g[i] + bb[i];
}

// =====================================================================
// corr[b,tau] = (1/D) * sum_t R[b, t, (t-tau)%L]  where R = Q K^T (from WMMA).
// (identical to irfft(rfft(q)*conj(rfft(k))) averaged over heads/channels)
// =====================================================================
__global__ void diag_corr_kernel(const float* __restrict__ R, float* __restrict__ corr) {
  const int tau = blockIdx.x, b = blockIdx.y, tid = threadIdx.x;
  const float* Rb = R + (size_t)b * L_ * L_;
  float s = 0.f;
  for (int t = tid; t < L_; t += 256) {
    int ss = (t - tau) & (L_ - 1);
    s += Rb[(size_t)t * L_ + ss];
  }
  __shared__ float red[256];
  red[tid] = s; __syncthreads();
  for (int st = 128; st > 0; st >>= 1) { if (tid < st) red[tid] += red[tid + st]; __syncthreads(); }
  if (tid == 0) corr[(size_t)b * L_ + tau] = red[0] * (1.0f / (float)D_);
}

// top-6 + softmax (tiny: serial per batch element)
__global__ void topk_kernel(const float* __restrict__ corr, float* __restrict__ wts,
                            int* __restrict__ dls) {
  const int b = blockIdx.x;
  if (threadIdx.x != 0) return;
  const float* c = corr + (size_t)b * L_;
  float vals[TOPK_]; int idxs[TOPK_];
  for (int kk = 0; kk < TOPK_; ++kk) {
    float best = -3.4e38f; int bi = 0;
    for (int t = 0; t < L_; ++t) {
      bool skip = false;
      for (int j = 0; j < kk; ++j) skip |= (idxs[j] == t);
      float v = c[t];
      if (!skip && v > best) { best = v; bi = t; }
    }
    vals[kk] = best; idxs[kk] = bi;
  }
  float mx = vals[0];
  for (int kk = 1; kk < TOPK_; ++kk) mx = fmaxf(mx, vals[kk]);
  float e[TOPK_], se = 0.f;
  for (int kk = 0; kk < TOPK_; ++kk) { e[kk] = expf(vals[kk] - mx); se += e[kk]; }
  for (int kk = 0; kk < TOPK_; ++kk) {
    wts[b * TOPK_ + kk] = e[kk] / se;
    dls[b * TOPK_ + kk] = idxs[kk];
  }
}

// out[b,l,d] = sum_k w[b,k] * v[b,(l+delay_k)%L, d]
__global__ void agg_kernel(const float* __restrict__ v, const float* __restrict__ wts,
                           const int* __restrict__ dls, float* __restrict__ out) {
  size_t i = (size_t)blockIdx.x * 256 + threadIdx.x;   // over B*L*D exactly
  int d = (int)(i & (D_ - 1));
  size_t r = i >> 9;
  int l = (int)(r & (L_ - 1));
  int b = (int)(r >> 9);
  const float* vb = v + (size_t)b * L_ * D_;
  float s = 0.f;
#pragma unroll
  for (int kk = 0; kk < TOPK_; ++kk) {
    int ls = (l + dls[b * TOPK_ + kk]) & (L_ - 1);
    s += wts[b * TOPK_ + kk] * vb[(size_t)ls * D_ + d];
  }
  out[i] = s;
}

__global__ void add_kernel(const float* __restrict__ a, const float* __restrict__ b2,
                           float* __restrict__ o) {
  size_t i = (size_t)blockIdx.x * 256 + threadIdx.x;
  o[i] = a[i] + b2[i];
}

// series decomposition residual: res = s - movavg_25(s) with edge replication
__global__ void decomp_kernel(const float* __restrict__ s, float* __restrict__ res) {
  size_t i = (size_t)blockIdx.x * 256 + threadIdx.x;   // over B*L*D exactly
  int d = (int)(i & (D_ - 1));
  size_t r = i >> 9;
  int l = (int)(r & (L_ - 1));
  int b = (int)(r >> 9);
  const float* sb = s + (size_t)b * L_ * D_;
  float sum = 0.f;
#pragma unroll
  for (int o = -PAD_; o <= PAD_; ++o) {
    int ls = l + o;
    ls = ls < 0 ? 0 : (ls > L_ - 1 ? L_ - 1 : ls);
    sum += sb[(size_t)ls * D_ + d];
  }
  res[i] = sb[(size_t)l * D_ + d] - sum * (1.0f / (float)MA_);
}

// rep[b,:] = [out0[b,L-1,:], out1[b,L-1,:], ie[b,:]]
__global__ void rep_kernel(const float* __restrict__ o0, const float* __restrict__ o1,
                           const float* __restrict__ ie, float* __restrict__ rep) {
  const int R = 2 * D_ + HS_;  // 1536
  int i = blockIdx.x * 256 + threadIdx.x;
  if (i >= B_ * R) return;
  int b = i / R, j = i % R;
  float v;
  if (j < D_)            v = o0[((size_t)b * L_ + (L_ - 1)) * D_ + j];
  else if (j < 2 * D_)   v = o1[((size_t)b * L_ + (L_ - 1)) * D_ + (j - D_)];
  else                   v = ie[(size_t)b * HS_ + (j - 2 * D_)];
  rep[i] = v;
}

// out[b] = h[b,:] . w + b2   (N=1 head)
__global__ void final_kernel(const float* __restrict__ h, const float* __restrict__ w,
                             const float* __restrict__ b2, float* __restrict__ out) {
  const int b = blockIdx.x, tid = threadIdx.x;
  const float* hb = h + (size_t)b * HS_;
  float s = 0.f;
  for (int i = tid; i < HS_; i += 256) s += hb[i] * w[i];
  __shared__ float red[256];
  red[tid] = s; __syncthreads();
  for (int st = 128; st > 0; st >>= 1) { if (tid < st) red[tid] += red[tid + st]; __syncthreads(); }
  if (tid == 0) out[b] = red[0] + b2[0];
}

// =====================================================================
extern "C" void kernel_launch(void* const* d_in, const int* in_sizes, int n_in,
                              void* d_out, int out_size, void* d_ws, size_t ws_size,
                              hipStream_t stream) {
  (void)in_sizes; (void)n_in; (void)out_size; (void)ws_size;

  const float* x_enc    = (const float*)d_in[0];
  const float* emb_w    = (const float*)d_in[1];
  const float* emb_b    = (const float*)d_in[2];
  const float* emb_g    = (const float*)d_in[3];
  const float* emb_beta = (const float*)d_in[4];
  const float* Wq       = (const float*)d_in[5];
  const float* bq       = (const float*)d_in[6];
  const float* Wk       = (const float*)d_in[7];
  const float* bk       = (const float*)d_in[8];
  const float* Wv       = (const float*)d_in[9];
  const float* bv       = (const float*)d_in[10];
  const float* Wo       = (const float*)d_in[11];
  const float* bo       = (const float*)d_in[12];
  const float* Wc1      = (const float*)d_in[13];
  const float* Wc2      = (const float*)d_in[14];
  const float* ip_w     = (const float*)d_in[15];
  const float* ip_b     = (const float*)d_in[16];
  const float* ip_g     = (const float*)d_in[17];
  const float* ip_beta  = (const float*)d_in[18];
  const float* rp_w1    = (const float*)d_in[19];
  const float* rp_b1    = (const float*)d_in[20];
  const float* rp_g     = (const float*)d_in[21];
  const float* rp_beta  = (const float*)d_in[22];
  const float* rp_w2    = (const float*)d_in[23];
  const float* rp_b2    = (const float*)d_in[24];
  float* out = (float*)d_out;

  const int BL = B_ * L_;
  const size_t BLD = (size_t)BL * D_;
  const size_t LD  = (size_t)L_ * D_;

  // workspace carving
  float* p    = (float*)d_ws;
  float* enc  = p; p += BLD;
  float* qb   = p; p += BLD;
  float* kb   = p; p += BLD;
  float* vb   = p; p += BLD;
  float* t2   = p; p += BLD;
  float* out0 = p; p += BLD;
  float* out1 = p; p += BLD;
  float* ffn  = p; p += (size_t)BL * DFF_;   // also reused as R = Q K^T scratch
  float* corr = p; p += (size_t)B_ * L_;
  float* wts  = p; p += B_ * TOPK_;
  int*   dls  = (int*)p; p += B_ * TOPK_;
  float* ieraw= p; p += B_ * HS_;
  float* ie   = p; p += B_ * HS_;
  float* rep  = p; p += B_ * (2 * D_ + HS_);
  float* hraw = p; p += B_ * HS_;
  float* hb   = p; p += B_ * HS_;

  const dim3 blk(256);
  auto ggrid = [](int M, int N) { return dim3((unsigned)((N + TN - 1) / TN),
                                              (unsigned)((M + TM - 1) / TM), 1u); };
  const int nElemBlocks = (int)(BLD / 256);   // B*L*D is an exact multiple of 256

  // ---- embedding + LN ----
  gemm_bf16_wmma<<<ggrid(BL, D_), blk, 0, stream>>>(x_enc, emb_w, emb_b, t2,
      BL, D_, C_, 0, 0, 0, 0, 0);
  ln_kernel<<<BL, 256, 0, stream>>>(t2, enc, emb_g, emb_beta, D_);

  // ---- encoder layers ----
  const float* xin = enc;
  float* outs[EL_] = {out0, out1};
  for (int i = 0; i < EL_; ++i) {
    const float* Wq_ = Wq + (size_t)i * D_ * D_;   const float* bq_ = bq + (size_t)i * D_;
    const float* Wk_ = Wk + (size_t)i * D_ * D_;   const float* bk_ = bk + (size_t)i * D_;
    const float* Wv_ = Wv + (size_t)i * D_ * D_;   const float* bv_ = bv + (size_t)i * D_;
    const float* Wo_ = Wo + (size_t)i * D_ * D_;   const float* bo_ = bo + (size_t)i * D_;
    const float* Wc1_ = Wc1 + (size_t)i * D_ * DFF_;
    const float* Wc2_ = Wc2 + (size_t)i * DFF_ * D_;

    gemm_bf16_wmma<<<ggrid(BL, D_), blk, 0, stream>>>(xin, Wq_, bq_, qb, BL, D_, D_, 0, 0, 0, 0, 0);
    gemm_bf16_wmma<<<ggrid(BL, D_), blk, 0, stream>>>(xin, Wk_, bk_, kb, BL, D_, D_, 0, 0, 0, 0, 0);
    gemm_bf16_wmma<<<ggrid(BL, D_), blk, 0, stream>>>(xin, Wv_, bv_, vb, BL, D_, D_, 0, 0, 0, 0, 0);

    // R[b] = Q[b] * K[b]^T  — batched WMMA GEMM (B^T mode), R -> ffn scratch
    gemm_bf16_wmma<<<dim3(L_ / TN, L_ / TM, B_), blk, 0, stream>>>(
        qb, kb, nullptr, ffn, L_, L_, D_, 0, 1, LD, LD, (size_t)L_ * L_);
    diag_corr_kernel<<<dim3(L_, B_), blk, 0, stream>>>(ffn, corr);
    topk_kernel<<<B_, 32, 0, stream>>>(corr, wts, dls);
    agg_kernel<<<nElemBlocks, blk, 0, stream>>>(vb, wts, dls, qb);         // attn out -> qb

    gemm_bf16_wmma<<<ggrid(BL, D_), blk, 0, stream>>>(qb, Wo_, bo_, kb, BL, D_, D_, 0, 0, 0, 0, 0);
    add_kernel<<<nElemBlocks, blk, 0, stream>>>(xin, kb, vb);              // s = x + new_x
    decomp_kernel<<<nElemBlocks, blk, 0, stream>>>(vb, t2);                // x' -> t2

    gemm_bf16_wmma<<<ggrid(BL, DFF_), blk, 0, stream>>>(t2, Wc1_, nullptr, ffn, BL, DFF_, D_, 1, 0, 0, 0, 0);
    gemm_bf16_wmma<<<ggrid(BL, D_), blk, 0, stream>>>(ffn, Wc2_, nullptr, qb, BL, D_, DFF_, 0, 0, 0, 0, 0);
    add_kernel<<<nElemBlocks, blk, 0, stream>>>(t2, qb, vb);               // x' + y
    decomp_kernel<<<nElemBlocks, blk, 0, stream>>>(vb, outs[i]);           // layer output

    xin = outs[i];
  }

  // ---- head ----
  gemm_bf16_wmma<<<ggrid(B_, HS_), blk, 0, stream>>>(x_enc, ip_w, ip_b, ieraw,
      B_, HS_, L_ * C_, 1, 0, 0, 0, 0);
  ln_kernel<<<B_, 256, 0, stream>>>(ieraw, ie, ip_g, ip_beta, HS_);
  rep_kernel<<<(B_ * (2 * D_ + HS_) + 255) / 256, blk, 0, stream>>>(out0, out1, ie, rep);
  gemm_bf16_wmma<<<ggrid(B_, HS_), blk, 0, stream>>>(rep, rp_w1, rp_b1, hraw,
      B_, HS_, EL_ * D_ + HS_, 1, 0, 0, 0, 0);
  ln_kernel<<<B_, 256, 0, stream>>>(hraw, hb, rp_g, rp_beta, HS_);
  final_kernel<<<B_, 256, 0, stream>>>(hb, rp_w2, rp_b2, out);
}
